// ResBlock_23321672418010
// MI455X (gfx1250) — matrix operand order repaired and checked by reading, not compiled, not run on hardware
//
#include <hip/hip_runtime.h>
#include <hip/hip_bf16.h>
#include <stdint.h>

// ---------------- problem constants ----------------
#define BSZ   8
#define LSEQ  8192
#define DMOD  128
#define HDIM  128
#define NTOK  (BSZ * LSEQ)        // 65536 tokens
#define CHUNK 128                 // scan chunk length
#define NCH   (LSEQ / CHUNK)      // 64 chunks per sequence

typedef float v8f  __attribute__((ext_vector_type(8)));
typedef __bf16 v16bf __attribute__((ext_vector_type(16)));
typedef unsigned int u32x4 __attribute__((ext_vector_type(4)));
typedef int i4v __attribute__((vector_size(16)));   // matches async-LDS builtin param

union AFrag { u32x4 q[2]; v16bf v; };

#if __has_builtin(__builtin_amdgcn_global_load_async_to_lds_b128)
#define USE_ASYNC_LDS 1
#else
#define USE_ASYNC_LDS 0
#endif

__device__ __forceinline__ unsigned short f2bf(float f) {
  unsigned u = __float_as_uint(f);
  u += 0x7FFFu + ((u >> 16) & 1u);          // round-to-nearest-even
  return (unsigned short)(u >> 16);
}

__device__ __forceinline__ v8f wmma_bf16(v16bf a, v16bf b, v8f c) {
  // emits v_wmma_f32_16x16x32_bf16 (CDNA5, wave32)
  return __builtin_amdgcn_wmma_f32_16x16x32_bf16(false, a, false, b, (short)0, c,
                                                 false, false);
}

// Stage packed-B fragments (KT*NT tiles of 1KB) from global into LDS, shared by
// all 8 waves of the block. Async load-to-LDS when the toolchain exposes it.
template <int KT, int NT>
__device__ __forceinline__ void stage_b(const unsigned short* __restrict__ Bp,
                                        int ntBase, int ntTotal,
                                        unsigned short* lB) {
  const int nchunk = KT * NT * 64;          // 16-byte chunks
  for (int c = threadIdx.x; c < nchunk; c += 256) {
    int tile = c >> 6, within = c & 63;
    int kt = tile / NT, nt = tile - kt * NT;
    const unsigned short* src =
        Bp + ((size_t)(kt * ntTotal + ntBase + nt) * 512 + within * 8);
    unsigned short* dst = lB + ((size_t)tile * 512 + within * 8);
#if USE_ASYNC_LDS
    __builtin_amdgcn_global_load_async_to_lds_b128(
        (__attribute__((address_space(1))) i4v*)src,
        (__attribute__((address_space(3))) i4v*)dst, 0, 0);
#else
    *(u32x4*)dst = *(const u32x4*)src;
#endif
  }
#if USE_ASYNC_LDS
#if __has_builtin(__builtin_amdgcn_s_wait_asynccnt)
  __builtin_amdgcn_s_wait_asynccnt(0);
#else
  asm volatile("s_wait_asynccnt 0" ::: "memory");
#endif
#endif
  __syncthreads();
}

// WMMA 16-row strip, A fragments preloaded once (reused by all N-tiles),
// N-tile outer loop -> one live accumulator -> room to pipeline B feeds.
// Epilogue fused per N-tile via functor (nt, acc).
template <int KT, int NT, class F>
__device__ __forceinline__ void gemm_run(const unsigned short* __restrict__ A,
                                         int ldk, int row0,
                                         const unsigned short* __restrict__ B,
                                         F epi) {
  const int lane = threadIdx.x & 31;
  const int mr = lane & 15, kh = lane >> 4;
  const unsigned short* arow = A + (size_t)(row0 + mr) * ldk;
  __builtin_prefetch(arow, 0, 3);
  AFrag af[KT];
#pragma unroll
  for (int kt = 0; kt < KT; ++kt) {
    // ISA 16-bit A 16x32 layout: elems 0-7 -> K = kh*8.., elems 8-15 -> K = 16+kh*8..
    af[kt].q[0] = *(const u32x4*)(arow + kt * 32 + kh * 8);
    af[kt].q[1] = *(const u32x4*)(arow + kt * 32 + 16 + kh * 8);
  }
#pragma unroll
  for (int nt = 0; nt < NT; ++nt) {
    v8f acc;
    for (int j = 0; j < 8; ++j) acc[j] = 0.f;
#pragma unroll
    for (int kt = 0; kt < KT; ++kt)
      acc = wmma_bf16(af[kt].v,
                      *(const v16bf*)(B + ((size_t)(kt * NT + nt) * 32 + lane) * 16),
                      acc);
    epi(nt, acc);
  }
}

// ---------------- tiny per-batch conditioning scalars ----------------
__global__ void scalars_kernel(const float* __restrict__ c, const float* w0,
                               const float* b0, const float* w1, const float* b1,
                               const float* w2, const float* b2, const float* w3,
                               const float* b3, const float* w4, const float* b4,
                               const float* w5, const float* b5,
                               float* __restrict__ sc) {
  int t = threadIdx.x;
  if (t >= 48) return;
  int p = t >> 3, bi = t & 7;
  const float* wsel[6] = {w0, w1, w2, w3, w4, w5};
  const float* bsel[6] = {b0, b1, b2, b3, b4, b5};
  const float* w = wsel[p];
  float s = bsel[p][0];
  for (int k = 0; k < DMOD; ++k) s += c[bi * DMOD + k] * w[k];
  sc[p * BSZ + bi] = s;   // [pair][batch]; pairs: s1,b1,g1,s2,b2,g2
}

// ---------------- pack fp32 weight (KxN row-major) into bf16 WMMA B-fragments
__global__ void pack_b_kernel(const float* __restrict__ W,
                              unsigned short* __restrict__ dst, int K, int N) {
  int pidx = blockIdx.x * blockDim.x + threadIdx.x;
  if (pidx >= K * N) return;
  int e = pidx & 15;
  int lane = (pidx >> 4) & 31;
  int rest = pidx >> 9;
  int NT = N >> 4;
  int nt = rest % NT;
  int kt = rest / NT;
  int n = nt * 16 + (lane & 15);
  int k = kt * 32 + (e >> 3) * 16 + (lane >> 4) * 8 + (e & 7);   // B 32x16: elems 0-7 K=kh*8.., elems 8-15 K=16+kh*8..
  dst[pidx] = f2bf(W[(size_t)k * N + n]);
}

// ---------------- row reductions for LayerNorm ----------------
__device__ __forceinline__ float block_sum128(float v, float* sm) {
  int t = threadIdx.x;
  sm[t] = v;
  __syncthreads();
  for (int s = 64; s > 0; s >>= 1) {
    if (t < s) sm[t] += sm[t + s];
    __syncthreads();
  }
  float r = sm[0];
  __syncthreads();
  return r;
}

// LN(x) -> affine by per-batch scalars -> LN again -> bf16 (rnn branch input)
__global__ void ln1_kernel(const float* __restrict__ x, const float* __restrict__ sc,
                           unsigned short* __restrict__ xln) {
  __shared__ float sm[128];
  int row = blockIdx.x, t = threadIdx.x;
  int bi = row >> 13;  // row / LSEQ
  float v = x[(size_t)row * DMOD + t];
  float mean = block_sum128(v, sm) * (1.f / 128.f);
  float d = v - mean;
  float var = block_sum128(d * d, sm) * (1.f / 128.f);
  float y = d * rsqrtf(var + 1e-6f);
  float h = (1.f + sc[0 * BSZ + bi]) * y + sc[1 * BSZ + bi];
  float m2 = block_sum128(h, sm) * (1.f / 128.f);
  float d2 = h - m2;
  float v2 = block_sum128(d2 * d2, sm) * (1.f / 128.f);
  xln[(size_t)row * DMOD + t] = f2bf(d2 * rsqrtf(v2 + 1e-6f));
}

// LN(x1) -> affine -> bf16 (mlp branch input)
__global__ void ln2_kernel(const float* __restrict__ x1, const float* __restrict__ sc,
                           unsigned short* __restrict__ h2) {
  __shared__ float sm[128];
  int row = blockIdx.x, t = threadIdx.x;
  int bi = row >> 13;
  float v = x1[(size_t)row * DMOD + t];
  float mean = block_sum128(v, sm) * (1.f / 128.f);
  float d = v - mean;
  float var = block_sum128(d * d, sm) * (1.f / 128.f);
  float y = d * rsqrtf(var + 1e-6f);
  h2[(size_t)row * DMOD + t] = f2bf((1.f + sc[3 * BSZ + bi]) * y + sc[4 * BSZ + bi]);
}

// ---------------- u = LN2(x) @ Win + b + pos ----------------
__global__ void gemm_u_kernel(const unsigned short* __restrict__ A,
                              const unsigned short* __restrict__ Bp,
                              const float* __restrict__ bias,
                              const float* __restrict__ pos,
                              float* __restrict__ uf, unsigned short* __restrict__ ub) {
  extern __shared__ unsigned short lB[];
  stage_b<4, 8>(Bp, 0, 8, lB);
  int wave = threadIdx.x >> 5, lane = threadIdx.x & 31;
  int row0 = (blockIdx.x * 8 + wave) * 16;
  int mr = lane & 15, kh = lane >> 4;
  gemm_run<4, 8>(A, 128, row0, lB, [&](int nt, v8f acc) {
    int col = nt * 16 + mr;
    for (int v = 0; v < 8; ++v) {
      int row = row0 + v + 8 * kh;
      int l = row & (LSEQ - 1);
      float u = acc[v] + bias[col] + pos[(size_t)l * HDIM + col];
      uf[(size_t)row * HDIM + col] = u;
      ub[(size_t)row * HDIM + col] = f2bf(u);
    }
  });
}

// ---------------- gates: gi/gr GEMMs + RG-LRU a_t, b_t ----------------
__global__ void gemm_gates_kernel(const unsigned short* __restrict__ ub,
                                  const unsigned short* __restrict__ Bi,
                                  const unsigned short* __restrict__ Br,
                                  const float* __restrict__ bi_b,
                                  const float* __restrict__ br_b,
                                  const float* __restrict__ a_param,
                                  const float* __restrict__ uf,
                                  float* __restrict__ at, float* __restrict__ bt) {
  extern __shared__ unsigned short lB[];
  unsigned short* lBi = lB;
  unsigned short* lBr = lB + 4 * 8 * 512;
  stage_b<4, 8>(Bi, 0, 8, lBi);
  stage_b<4, 8>(Br, 0, 8, lBr);
  int wave = threadIdx.x >> 5, lane = threadIdx.x & 31;
  int row0 = (blockIdx.x * 8 + wave) * 16;
  int mr = lane & 15, kh = lane >> 4;
  const unsigned short* arow = ub + (size_t)(row0 + mr) * 128;
  AFrag af[4];
#pragma unroll
  for (int kt = 0; kt < 4; ++kt) {
    af[kt].q[0] = *(const u32x4*)(arow + kt * 32 + kh * 8);
    af[kt].q[1] = *(const u32x4*)(arow + kt * 32 + 16 + kh * 8);
  }
#pragma unroll
  for (int nt = 0; nt < 8; ++nt) {
    v8f accI, accR;
    for (int j = 0; j < 8; ++j) { accI[j] = 0.f; accR[j] = 0.f; }
#pragma unroll
    for (int kt = 0; kt < 4; ++kt)
      accI = wmma_bf16(af[kt].v,
                       *(const v16bf*)(lBi + ((size_t)(kt * 8 + nt) * 32 + lane) * 16),
                       accI);
#pragma unroll
    for (int kt = 0; kt < 4; ++kt)
      accR = wmma_bf16(af[kt].v,
                       *(const v16bf*)(lBr + ((size_t)(kt * 8 + nt) * 32 + lane) * 16),
                       accR);
    int col = nt * 16 + mr;
    float la = __logf(a_param[col]);
    for (int v = 0; v < 8; ++v) {
      int row = row0 + v + 8 * kh;
      size_t idx = (size_t)row * HDIM + col;
      float gi = 1.f / (1.f + __expf(-(accI[v] + bi_b[col])));
      float gr = 1.f / (1.f + __expf(-(accR[v] + br_b[col])));
      float a_t = __expf(8.f * gr * la);      // a^(c*r_t)
      float u = uf[idx];
      float b_t = sqrtf(fmaxf(1.f - a_t * a_t, 0.f)) * gi * u;
      at[idx] = a_t;
      bt[idx] = b_t;
    }
  }
}

// ---------------- chunked bidirectional linear scan ----------------
__global__ void scan_partial_kernel(const float* __restrict__ at,
                                    const float* __restrict__ bt,
                                    float* __restrict__ fA, float* __restrict__ fB,
                                    float* __restrict__ bA, float* __restrict__ bB) {
  int blk = blockIdx.x;
  int bi = blk >> 6, cn = blk & 63;
  int ch = threadIdx.x;
  size_t base = ((size_t)bi * LSEQ + (size_t)cn * CHUNK) * HDIM + ch;
  float A = 1.f, B = 0.f;
  for (int t = 0; t < CHUNK; ++t) {
    float a = at[base + (size_t)t * HDIM], b = bt[base + (size_t)t * HDIM];
    B = a * B + b;
    A *= a;
  }
  fA[(size_t)blk * HDIM + ch] = A;
  fB[(size_t)blk * HDIM + ch] = B;
  float A2 = 1.f, B2 = 0.f;
  for (int t = CHUNK - 1; t >= 0; --t) {
    float a = at[base + (size_t)t * HDIM], b = bt[base + (size_t)t * HDIM];
    B2 = a * B2 + b;
    A2 *= a;
  }
  bA[(size_t)blk * HDIM + ch] = A2;
  bB[(size_t)blk * HDIM + ch] = B2;
}

__global__ void scan_carry_kernel(float* __restrict__ fA, float* __restrict__ fB,
                                  float* __restrict__ bA, float* __restrict__ bB) {
  int bi = blockIdx.x, ch = threadIdx.x;
  float h = 0.f;
  for (int cn = 0; cn < NCH; ++cn) {
    size_t idx = ((size_t)bi * NCH + cn) * HDIM + ch;
    float A = fA[idx], B = fB[idx];
    float pre = h;
    h = A * h + B;
    fB[idx] = pre;                 // carry entering chunk cn (forward)
  }
  float h2 = 0.f;
  for (int cn = NCH - 1; cn >= 0; --cn) {
    size_t idx = ((size_t)bi * NCH + cn) * HDIM + ch;
    float A = bA[idx], B = bB[idx];
    float pre = h2;
    h2 = A * h2 + B;
    bB[idx] = pre;                 // carry entering chunk cn (backward)
  }
}

__global__ void scan_apply_kernel(const float* __restrict__ at,
                                  const float* __restrict__ bt,
                                  const float* __restrict__ fB,
                                  const float* __restrict__ bB,
                                  unsigned short* __restrict__ cc) {
  int blk = blockIdx.x;
  int bi = blk >> 6, cn = blk & 63;
  int ch = threadIdx.x;
  size_t base = ((size_t)bi * LSEQ + (size_t)cn * CHUNK) * HDIM + ch;
  int tok0 = bi * LSEQ + cn * CHUNK;
  float h = fB[(size_t)blk * HDIM + ch];
  for (int t = 0; t < CHUNK; ++t) {
    size_t idx = base + (size_t)t * HDIM;
    h = at[idx] * h + bt[idx];
    cc[(size_t)(tok0 + t) * 256 + ch] = f2bf(h);
  }
  float h2 = bB[(size_t)blk * HDIM + ch];
  for (int t = CHUNK - 1; t >= 0; --t) {
    size_t idx = base + (size_t)t * HDIM;
    h2 = at[idx] * h2 + bt[idx];
    cc[(size_t)(tok0 + t) * 256 + 128 + ch] = f2bf(h2);
  }
}

// ---------------- r = concat @ Wout + b ; x1 = x + g1*r ----------------
__global__ void gemm_r_kernel(const unsigned short* __restrict__ cc,
                              const unsigned short* __restrict__ Bp,
                              const float* __restrict__ bias,
                              const float* __restrict__ x,
                              const float* __restrict__ sc, float* __restrict__ x1) {
  extern __shared__ unsigned short lB[];
  stage_b<8, 8>(Bp, 0, 8, lB);
  int wave = threadIdx.x >> 5, lane = threadIdx.x & 31;
  int row0 = (blockIdx.x * 8 + wave) * 16;
  int mr = lane & 15, kh = lane >> 4;
  gemm_run<8, 8>(cc, 256, row0, lB, [&](int nt, v8f acc) {
    int col = nt * 16 + mr;
    for (int v = 0; v < 8; ++v) {
      int row = row0 + v + 8 * kh;
      int bi = row >> 13;
      float g1 = sc[2 * BSZ + bi];
      size_t idx = (size_t)row * DMOD + col;
      x1[idx] = x[idx] + g1 * (acc[v] + bias[col]);
    }
  });
}

// ---------------- m1 = gelu(h2 @ W1 + b1) ----------------
__device__ __forceinline__ float gelu_tanh(float x) {
  float x3 = x * x * x;
  return 0.5f * x * (1.f + tanhf(0.7978845608028654f * (x + 0.044715f * x3)));
}

__global__ void gemm_m1_kernel(const unsigned short* __restrict__ h2,
                               const unsigned short* __restrict__ Bp,
                               const float* __restrict__ bias,
                               unsigned short* __restrict__ m1) {
  extern __shared__ unsigned short lB[];
  int ntBase = blockIdx.y * 8;             // N=512 -> 32 N-tiles, 4 groups of 8
  stage_b<4, 8>(Bp, ntBase, 32, lB);
  int wave = threadIdx.x >> 5, lane = threadIdx.x & 31;
  int row0 = (blockIdx.x * 8 + wave) * 16;
  int mr = lane & 15, kh = lane >> 4;
  gemm_run<4, 8>(h2, 128, row0, lB, [&](int nt, v8f acc) {
    int col = (ntBase + nt) * 16 + mr;
    for (int v = 0; v < 8; ++v) {
      int row = row0 + v + 8 * kh;
      m1[(size_t)row * 512 + col] = f2bf(gelu_tanh(acc[v] + bias[col]));
    }
  });
}

// ---------------- out = x1 + g2*(m1 @ W2 + b2) ----------------
__global__ void gemm_m2_kernel(const unsigned short* __restrict__ m1,
                               const unsigned short* __restrict__ Bp,
                               const float* __restrict__ bias,
                               const float* __restrict__ x1,
                               const float* __restrict__ sc, float* __restrict__ out) {
  // packed B is 128KB here -> read straight from global (L2-resident weights)
  int wave = threadIdx.x >> 5, lane = threadIdx.x & 31;
  int row0 = (blockIdx.x * 8 + wave) * 16;
  int mr = lane & 15, kh = lane >> 4;
  gemm_run<16, 8>(m1, 512, row0, Bp, [&](int nt, v8f acc) {
    int col = nt * 16 + mr;
    for (int v = 0; v < 8; ++v) {
      int row = row0 + v + 8 * kh;
      int bi = row >> 13;
      float g2 = sc[5 * BSZ + bi];
      size_t idx = (size_t)row * DMOD + col;
      out[idx] = x1[idx] + g2 * (acc[v] + bias[col]);
    }
  });
}

// ---------------- host side ----------------
extern "C" void kernel_launch(void* const* d_in, const int* in_sizes, int n_in,
                              void* d_out, int out_size, void* d_ws, size_t ws_size,
                              hipStream_t stream) {
  const float* x       = (const float*)d_in[0];
  const float* c       = (const float*)d_in[1];
  const float* cln1_sw = (const float*)d_in[2];
  const float* cln1_sb = (const float*)d_in[3];
  const float* cln1_bw = (const float*)d_in[4];
  const float* cln1_bb = (const float*)d_in[5];
  const float* gate1_w = (const float*)d_in[6];
  const float* gate1_b = (const float*)d_in[7];
  const float* rnn_in_w = (const float*)d_in[8];
  const float* rnn_in_b = (const float*)d_in[9];
  const float* pos_emb  = (const float*)d_in[10];
  const float* rnn_wi   = (const float*)d_in[11];
  const float* rnn_bi   = (const float*)d_in[12];
  const float* rnn_wr   = (const float*)d_in[13];
  const float* rnn_br   = (const float*)d_in[14];
  const float* rnn_a    = (const float*)d_in[15];
  const float* rnn_out_w = (const float*)d_in[16];
  const float* rnn_out_b = (const float*)d_in[17];
  const float* cln2_sw = (const float*)d_in[18];
  const float* cln2_sb = (const float*)d_in[19];
  const float* cln2_bw = (const float*)d_in[20];
  const float* cln2_bb = (const float*)d_in[21];
  const float* gate2_w = (const float*)d_in[22];
  const float* gate2_b = (const float*)d_in[23];
  const float* mlp_w1  = (const float*)d_in[24];
  const float* mlp_b1  = (const float*)d_in[25];
  const float* mlp_w2  = (const float*)d_in[26];
  const float* mlp_b2  = (const float*)d_in[27];
  float* out = (float*)d_out;

  // workspace layout (bytes), peak ~170 MB with region reuse
  char* ws = (char*)d_ws;
  size_t off = 0;
  auto alloc = [&](size_t bytes) {
    size_t o = off;
    off += (bytes + 255) & ~(size_t)255;
    return o;
  };
  size_t SC  = alloc(256);                        // 6*8 scalars
  size_t PKI = alloc((size_t)128 * 128 * 2);      // rnn_in_w packed
  size_t PWI = alloc((size_t)128 * 128 * 2);      // rnn_wi
  size_t PWR = alloc((size_t)128 * 128 * 2);      // rnn_wr
  size_t PWO = alloc((size_t)256 * 128 * 2);      // rnn_out_w
  size_t PW1 = alloc((size_t)128 * 512 * 2);      // mlp_w1
  size_t PW2 = alloc((size_t)512 * 128 * 2);      // mlp_w2
  size_t FA  = alloc((size_t)BSZ * NCH * HDIM * 4);
  size_t FB  = alloc((size_t)BSZ * NCH * HDIM * 4);
  size_t BA  = alloc((size_t)BSZ * NCH * HDIM * 4);
  size_t BB  = alloc((size_t)BSZ * NCH * HDIM * 4);
  size_t XLN = alloc((size_t)NTOK * 128 * 2);     // later reused as h2
  size_t UF  = alloc((size_t)NTOK * 128 * 4);     // later reused as concat (same bytes)
  size_t UB  = alloc((size_t)NTOK * 128 * 2);
  size_t X1  = alloc((size_t)NTOK * 128 * 4);
  size_t AT  = alloc((size_t)NTOK * 128 * 4);     // AT..BT later reused as m1 (bf16, 64MB)
  size_t BT  = alloc((size_t)NTOK * 128 * 4);
  size_t H2 = XLN, CC = UF, M1 = AT;
  (void)ws_size; (void)in_sizes; (void)n_in; (void)out_size;

  float* sc = (float*)(ws + SC);

  scalars_kernel<<<1, 64, 0, stream>>>(c, cln1_sw, cln1_sb, cln1_bw, cln1_bb,
                                       gate1_w, gate1_b, cln2_sw, cln2_sb, cln2_bw,
                                       cln2_bb, gate2_w, gate2_b, sc);

  pack_b_kernel<<<(128 * 128) / 256, 256, 0, stream>>>(rnn_in_w, (unsigned short*)(ws + PKI), 128, 128);
  pack_b_kernel<<<(128 * 128) / 256, 256, 0, stream>>>(rnn_wi, (unsigned short*)(ws + PWI), 128, 128);
  pack_b_kernel<<<(128 * 128) / 256, 256, 0, stream>>>(rnn_wr, (unsigned short*)(ws + PWR), 128, 128);
  pack_b_kernel<<<(256 * 128) / 256, 256, 0, stream>>>(rnn_out_w, (unsigned short*)(ws + PWO), 256, 128);
  pack_b_kernel<<<(128 * 512) / 256, 256, 0, stream>>>(mlp_w1, (unsigned short*)(ws + PW1), 128, 512);
  pack_b_kernel<<<(512 * 128) / 256, 256, 0, stream>>>(mlp_w2, (unsigned short*)(ws + PW2), 512, 128);

  ln1_kernel<<<NTOK, 128, 0, stream>>>(x, sc, (unsigned short*)(ws + XLN));

  const int GEMM_BLOCKS = NTOK / (16 * 8);  // 512 blocks, 8 waves * 16 rows each
  gemm_u_kernel<<<GEMM_BLOCKS, 256, 32768, stream>>>(
      (const unsigned short*)(ws + XLN), (const unsigned short*)(ws + PKI),
      rnn_in_b, pos_emb, (float*)(ws + UF), (unsigned short*)(ws + UB));

  gemm_gates_kernel<<<GEMM_BLOCKS, 256, 65536, stream>>>(
      (const unsigned short*)(ws + UB), (const unsigned short*)(ws + PWI),
      (const unsigned short*)(ws + PWR), rnn_bi, rnn_br, rnn_a,
      (const float*)(ws + UF), (float*)(ws + AT), (float*)(ws + BT));

  scan_partial_kernel<<<BSZ * NCH, 128, 0, stream>>>(
      (const float*)(ws + AT), (const float*)(ws + BT), (float*)(ws + FA),
      (float*)(ws + FB), (float*)(ws + BA), (float*)(ws + BB));
  scan_carry_kernel<<<BSZ, 128, 0, stream>>>((float*)(ws + FA), (float*)(ws + FB),
                                             (float*)(ws + BA), (float*)(ws + BB));
  scan_apply_kernel<<<BSZ * NCH, 128, 0, stream>>>(
      (const float*)(ws + AT), (const float*)(ws + BT), (const float*)(ws + FB),
      (const float*)(ws + BB), (unsigned short*)(ws + CC));

  gemm_r_kernel<<<GEMM_BLOCKS, 256, 65536, stream>>>(
      (const unsigned short*)(ws + CC), (const unsigned short*)(ws + PWO),
      rnn_out_b, x, sc, (float*)(ws + X1));

  ln2_kernel<<<NTOK, 128, 0, stream>>>((const float*)(ws + X1), sc,
                                       (unsigned short*)(ws + H2));

  gemm_m1_kernel<<<dim3(GEMM_BLOCKS, 4), 256, 32768, stream>>>(
      (const unsigned short*)(ws + H2), (const unsigned short*)(ws + PW1), mlp_b1,
      (unsigned short*)(ws + M1));

  gemm_m2_kernel<<<GEMM_BLOCKS, 256, 0, stream>>>(
      (const unsigned short*)(ws + M1), (const unsigned short*)(ws + PW2), mlp_b2,
      (const float*)(ws + X1), sc, out);
}
